// WindowAttention_11605001634473
// MI455X (gfx1250) — matrix operand order repaired and checked by reading, not compiled, not run on hardware
//
#include <hip/hip_runtime.h>
#include <hip/hip_bf16.h>
#include <math.h>

#define DIM 384
#define NHEAD 12
#define HD 32
#define NTOK 64
#define NWIN 1024
#define CPB_HID 512
#define TABLE 225
#define MAX_LOGIT 4.60517018598809136804f   // log(100)

typedef __bf16 bf16;
typedef __bf16 v16bf __attribute__((ext_vector_type(16)));
typedef __bf16 v8bf  __attribute__((ext_vector_type(8)));
typedef __bf16 v4bf  __attribute__((ext_vector_type(4)));
typedef float  v8f   __attribute__((ext_vector_type(8)));

// ---- WMMA operand loaders (CDNA5 §7.12.2 layouts, bf16, K=32 per step) ----
// A (MxK): lane l(0..15)=row, elems 0..7 = K k0..k0+7, elems 8..15 = k0+16..23;
//          lane l+16 = same row, K +8 / +24.  Two 16B chunks per lane.
static __device__ __forceinline__ v16bf load_tile_a(const bf16* base, int row0,
                                                    int ldm, int k0, int lane) {
  const int r  = lane & 15;
  const int ko = (lane >> 4) << 3;            // 0 or 8
  const bf16* p = base + (row0 + r) * ldm + k0 + ko;
  v8bf lo = *(const v8bf*)(p);
  v8bf hi = *(const v8bf*)(p + 16);
  return __builtin_shufflevector(lo, hi, 0,1,2,3,4,5,6,7,8,9,10,11,12,13,14,15);
}
// B (KxN) supplied as N-major (col, K): lane l(0..15)=col, elems 0..7 = K k0..k0+7,
//          elems 8..15 = k0+16..23; lane l+16 = same col, K +8 / +24.  Two 16B chunks per lane.
static __device__ __forceinline__ v16bf load_tile_b(const bf16* base, int col0,
                                                    int ldm, int k0, int lane) {
  const int c  = lane & 15;
  const int ko = (lane >> 4) << 3;            // 0 or 8
  const bf16* p = base + (col0 + c) * ldm + k0 + ko;
  v8bf lo = *(const v8bf*)(p);
  v8bf hi = *(const v8bf*)(p + 16);
  return __builtin_shufflevector(lo, hi, 0,1,2,3,4,5,6,7,8,9,10,11,12,13,14,15);
}

static __device__ __forceinline__ v8f wmma_bf(v16bf a, v16bf b, v8f c) {
  return __builtin_amdgcn_wmma_f32_16x16x32_bf16(false, a, false, b, (short)0, c,
                                                 false, false);
}

// ---------------- prep kernel 1: CPB MLP (225 x 2 -> 512 relu -> 12) --------
__global__ __launch_bounds__(256)
void cpb_mlp_kernel(const float* __restrict__ table, const float* __restrict__ w1,
                    const float* __restrict__ b1, const float* __restrict__ w2,
                    const float* __restrict__ b2, float* __restrict__ rpb_head) {
  __shared__ float red[256 * NHEAD];
  const int t = blockIdx.x, tid = threadIdx.x;
  const float t0 = table[t * 2 + 0], t1 = table[t * 2 + 1];
  float acc[NHEAD];
#pragma unroll
  for (int h = 0; h < NHEAD; ++h) acc[h] = 0.f;
  for (int j = tid; j < CPB_HID; j += 256) {
    const float hv = fmaxf(0.f, t0 * w1[j] + t1 * w1[CPB_HID + j] + b1[j]);
#pragma unroll
    for (int h = 0; h < NHEAD; ++h) acc[h] += hv * w2[j * NHEAD + h];
  }
#pragma unroll
  for (int h = 0; h < NHEAD; ++h) red[tid * NHEAD + h] = acc[h];
  __syncthreads();
  for (int s = 128; s > 0; s >>= 1) {
    if (tid < s)
      for (int h = 0; h < NHEAD; ++h)
        red[tid * NHEAD + h] += red[(tid + s) * NHEAD + h];
    __syncthreads();
  }
  if (tid < NHEAD) rpb_head[t * NHEAD + tid] = red[tid] + b2[tid];
}

// ---------------- prep kernel 2: gather + 16*sigmoid -> (H,64,64) -----------
__global__ __launch_bounds__(256)
void bias_gather_kernel(const float* __restrict__ rpb_head,
                        const int* __restrict__ rpb_idx,
                        float* __restrict__ bias16) {
  const int e = blockIdx.x * 256 + threadIdx.x;
  if (e >= NHEAD * NTOK * NTOK) return;
  const int h = e >> 12, ij = e & 4095;
  const float v = rpb_head[rpb_idx[ij] * NHEAD + h];
  bias16[e] = 16.f / (1.f + __expf(-v));
}

// ---------------- prep kernel 3: transpose+convert weights to bf16 ----------
__global__ __launch_bounds__(256)
void prep_weights_kernel(const float* __restrict__ qkv_w,
                         const float* __restrict__ proj_w,
                         bf16* __restrict__ wqkv_t, bf16* __restrict__ wproj_t) {
  const int i = blockIdx.x * 256 + threadIdx.x;
  const int n_qkv = 3 * DIM * DIM;            // 1152*384
  if (i < n_qkv) {
    const int o = i / DIM, k = i % DIM;       // (out, k) row-major
    wqkv_t[i] = (bf16)qkv_w[k * (3 * DIM) + o];
  } else {
    const int j = i - n_qkv;
    if (j < DIM * DIM) {
      const int o = j / DIM, k = j % DIM;
      wproj_t[j] = (bf16)proj_w[k * DIM + o];
    }
  }
}

// ---------------- fused per-window attention + projection -------------------
__global__ __launch_bounds__(256)
void swin_attn_kernel(const float* __restrict__ x, const float* __restrict__ mask,
                      const float* __restrict__ logit_scale,
                      const float* __restrict__ proj_b,
                      const float* __restrict__ bias16,
                      const bf16* __restrict__ wqkv_t,
                      const bf16* __restrict__ wproj_t, float* __restrict__ out) {
  __shared__ bf16  s_x[NTOK * DIM];     // x window, bf16        48 KB
  __shared__ bf16  s_ctx[NTOK * DIM];   // per-window context    48 KB
  __shared__ bf16  s_qn[NTOK * HD];     // normalized q           4 KB
  __shared__ bf16  s_kn[NTOK * HD];     // normalized k           4 KB
  __shared__ bf16  s_vt[HD * NTOK];     // v transposed (d,key)   4 KB
  __shared__ float s_attn[NTOK * NTOK]; // fp32 logits           16 KB
  __shared__ bf16  s_p[NTOK * NTOK];    // softmax probs bf16     8 KB

  const int b = blockIdx.x;
  const int tid = threadIdx.x;
  const int w = tid >> 5, lane = tid & 31;
  const float* xw = x + (size_t)b * NTOK * DIM;
  const float* mw = mask + (size_t)(b & 63) * NTOK * NTOK;

  // ---- ingest: float4 global loads -> packed bf16 -> ds_store_b64 ----------
  for (int i = tid; i < (NTOK * DIM) / 4; i += 256) {
    const float4 f = ((const float4*)xw)[i];
    v4bf r;
    r[0] = (bf16)f.x; r[1] = (bf16)f.y; r[2] = (bf16)f.z; r[3] = (bf16)f.w;
    *(v4bf*)(s_x + i * 4) = r;
  }
  __syncthreads();

  const int roff = (lane >> 4) << 3;          // C/D row offset per half-wave
  const int cl   = lane & 15;                 // C/D column within tile

  for (int h = 0; h < NHEAD; ++h) {
    const float scale = __expf(fminf(logit_scale[h], MAX_LOGIT));

    // ---- phase 1: q/k/v = x @ W_head; wave w owns tile (mi,ni) of all 3 ----
    {
      const int mi = w >> 1, ni = w & 1;
      const int o0 = h * HD + ni * 16;        // q col base; k=+DIM, v=+2*DIM
      v8f aq = {}, ak = {}, av = {};
      for (int kk = 0; kk < DIM; kk += 32) {
        v16bf a  = load_tile_a(s_x, mi * 16, DIM, kk, lane);
        v16bf bq = load_tile_b(wqkv_t, o0, DIM, kk, lane);
        v16bf bk = load_tile_b(wqkv_t, o0 + DIM, DIM, kk, lane);
        v16bf bv = load_tile_b(wqkv_t, o0 + 2 * DIM, DIM, kk, lane);
        aq = wmma_bf(a, bq, aq);
        ak = wmma_bf(a, bk, ak);
        av = wmma_bf(a, bv, av);
      }
      const int cc = ni * 16 + cl;
#pragma unroll
      for (int vr = 0; vr < 8; ++vr) {
        const int rr = mi * 16 + roff + vr;
        s_qn[rr * HD + cc]   = (bf16)aq[vr];
        s_kn[rr * HD + cc]   = (bf16)ak[vr];
        s_vt[cc * NTOK + rr] = (bf16)av[vr];  // transposed
      }
    }
    __syncthreads();

    // ---- phase 1.5: cosine-normalize q and k rows --------------------------
    if (tid < 128) {
      bf16* rowp = (tid < 64) ? (s_qn + tid * HD) : (s_kn + (tid - 64) * HD);
      float ssq = 0.f;
#pragma unroll
      for (int d = 0; d < HD; ++d) { const float fv = (float)rowp[d]; ssq += fv * fv; }
      const float inv = rsqrtf(fmaxf(ssq, 1e-12f));
#pragma unroll
      for (int d = 0; d < HD; ++d) rowp[d] = (bf16)((float)rowp[d] * inv);
    }
    __syncthreads();

    // ---- phase 2: logits; wave w owns tiles (mi0,ni2),(mi0+2,ni2), shared B -
    {
      const int ni2 = w & 3, mi0 = w >> 2;
      v16bf bb = load_tile_b(s_kn, ni2 * 16, HD, 0, lane);
      v16bf a0 = load_tile_a(s_qn, mi0 * 16, HD, 0, lane);
      v16bf a1 = load_tile_a(s_qn, (mi0 + 2) * 16, HD, 0, lane);
      v8f z = {};
      v8f acc0 = wmma_bf(a0, bb, z);
      v8f acc1 = wmma_bf(a1, bb, z);
      const int cc = ni2 * 16 + cl;
#pragma unroll
      for (int vr = 0; vr < 8; ++vr) {
        const int r0 = mi0 * 16 + roff + vr;
        const int r1 = r0 + 32;
        s_attn[r0 * NTOK + cc] = scale * acc0[vr]
                               + bias16[h * 4096 + r0 * NTOK + cc]
                               + mw[r0 * NTOK + cc];
        s_attn[r1 * NTOK + cc] = scale * acc1[vr]
                               + bias16[h * 4096 + r1 * NTOK + cc]
                               + mw[r1 * NTOK + cc];
      }
    }
    __syncthreads();

    // ---- phase 3: row softmax, 4 lanes per row, xor-butterfly reduce -------
    {
      const int row = tid >> 2;               // 0..63
      const int seg = tid & 3;                // 16-col segment
      const float* rp = s_attn + row * NTOK + seg * 16;
      float vals[16];
#pragma unroll
      for (int j = 0; j < 16; ++j) vals[j] = rp[j];
      float mx = vals[0];
#pragma unroll
      for (int j = 1; j < 16; ++j) mx = fmaxf(mx, vals[j]);
      mx = fmaxf(mx, __shfl_xor(mx, 1, 32));
      mx = fmaxf(mx, __shfl_xor(mx, 2, 32));
      float sm = 0.f;
#pragma unroll
      for (int j = 0; j < 16; ++j) { vals[j] = __expf(vals[j] - mx); sm += vals[j]; }
      sm += __shfl_xor(sm, 1, 32);
      sm += __shfl_xor(sm, 2, 32);
      const float invs = 1.f / sm;
      v4bf* pp = (v4bf*)(s_p + row * NTOK + seg * 16);
#pragma unroll
      for (int j4 = 0; j4 < 4; ++j4) {
        v4bf r;
        r[0] = (bf16)(vals[j4 * 4 + 0] * invs);
        r[1] = (bf16)(vals[j4 * 4 + 1] * invs);
        r[2] = (bf16)(vals[j4 * 4 + 2] * invs);
        r[3] = (bf16)(vals[j4 * 4 + 3] * invs);
        pp[j4] = r;
      }
    }
    __syncthreads();

    // ---- phase 4: ctx_h = P @ V  (8 tiles, K=64) ---------------------------
    {
      const int mi = w >> 1, ni = w & 1;
      v8f acc = {};
      for (int kk = 0; kk < NTOK; kk += 32) {
        v16bf a  = load_tile_a(s_p, mi * 16, NTOK, kk, lane);
        v16bf bb = load_tile_b(s_vt, ni * 16, NTOK, kk, lane);
        acc = wmma_bf(a, bb, acc);
      }
      const int cc = ni * 16 + cl;
#pragma unroll
      for (int vr = 0; vr < 8; ++vr) {
        const int rr = mi * 16 + roff + vr;
        s_ctx[rr * DIM + h * HD + cc] = (bf16)acc[vr];
      }
    }
    __syncthreads();
  }

  // ---- phase 5: out = ctx @ proj + b; wave w: mi fixed, 12 ni in 3 groups ----
  float* ow = out + (size_t)b * NTOK * DIM;
  {
    const int mi = w & 3;
    const int nb = w >> 2;                    // 0 or 1
    for (int g = 0; g < 3; ++g) {
      const int ni0 = nb + 2 * (4 * g + 0);
      const int ni1 = nb + 2 * (4 * g + 1);
      const int ni2 = nb + 2 * (4 * g + 2);
      const int ni3 = nb + 2 * (4 * g + 3);
      v8f c0 = {}, c1 = {}, c2 = {}, c3 = {};
      for (int kk = 0; kk < DIM; kk += 32) {
        v16bf a  = load_tile_a(s_ctx, mi * 16, DIM, kk, lane);
        v16bf b0 = load_tile_b(wproj_t, ni0 * 16, DIM, kk, lane);
        v16bf b1 = load_tile_b(wproj_t, ni1 * 16, DIM, kk, lane);
        v16bf b2 = load_tile_b(wproj_t, ni2 * 16, DIM, kk, lane);
        v16bf b3 = load_tile_b(wproj_t, ni3 * 16, DIM, kk, lane);
        c0 = wmma_bf(a, b0, c0);
        c1 = wmma_bf(a, b1, c1);
        c2 = wmma_bf(a, b2, c2);
        c3 = wmma_bf(a, b3, c3);
      }
      const float pb0 = proj_b[ni0 * 16 + cl];
      const float pb1 = proj_b[ni1 * 16 + cl];
      const float pb2 = proj_b[ni2 * 16 + cl];
      const float pb3 = proj_b[ni3 * 16 + cl];
#pragma unroll
      for (int vr = 0; vr < 8; ++vr) {
        const int rr = mi * 16 + roff + vr;
        ow[rr * DIM + ni0 * 16 + cl] = c0[vr] + pb0;
        ow[rr * DIM + ni1 * 16 + cl] = c1[vr] + pb1;
        ow[rr * DIM + ni2 * 16 + cl] = c2[vr] + pb2;
        ow[rr * DIM + ni3 * 16 + cl] = c3[vr] + pb3;
      }
    }
  }
}

extern "C" void kernel_launch(void* const* d_in, const int* in_sizes, int n_in,
                              void* d_out, int out_size, void* d_ws, size_t ws_size,
                              hipStream_t stream) {
  (void)in_sizes; (void)n_in; (void)out_size; (void)ws_size;
  const float* x           = (const float*)d_in[0];
  const float* mask        = (const float*)d_in[1];
  const float* qkv_w       = (const float*)d_in[2];
  const float* proj_w      = (const float*)d_in[3];
  const float* proj_b      = (const float*)d_in[4];
  const float* cpb_w1      = (const float*)d_in[5];
  const float* cpb_b1      = (const float*)d_in[6];
  const float* cpb_w2      = (const float*)d_in[7];
  const float* cpb_b2      = (const float*)d_in[8];
  const float* logit_scale = (const float*)d_in[9];
  const float* rpb_table   = (const float*)d_in[10];
  const int*   rpb_idx     = (const int*)d_in[11];

  char* ws = (char*)d_ws;
  float* rpb_head = (float*)(ws);                               // 10.8 KB
  float* bias16   = (float*)(ws + 16384);                       // 192 KB
  bf16*  wqkv_t   = (bf16*)(ws + 16384 + 196608);               // 864 KB
  bf16*  wproj_t  = (bf16*)(ws + 16384 + 196608 + 884736);      // 288 KB

  cpb_mlp_kernel<<<TABLE, 256, 0, stream>>>(rpb_table, cpb_w1, cpb_b1, cpb_w2,
                                            cpb_b2, rpb_head);
  bias_gather_kernel<<<(NHEAD * NTOK * NTOK + 255) / 256, 256, 0, stream>>>(
      rpb_head, rpb_idx, bias16);
  prep_weights_kernel<<<(3 * DIM * DIM + DIM * DIM + 255) / 256, 256, 0, stream>>>(
      qkv_w, proj_w, wqkv_t, wproj_t);
  swin_attn_kernel<<<NWIN, 256, 0, stream>>>(x, mask, logit_scale, proj_b, bias16,
                                             wqkv_t, wproj_t, (float*)d_out);
}
